// SpGAT_1_1_86887188398709
// MI455X (gfx1250) — compile-verified, matrix operand order checked by reading, not run on hardware
//
#include <hip/hip_runtime.h>
#include <hip/hip_bf16.h>

// ---------------------------------------------------------------------------
// CDNA5 (gfx1250) SpGAT pipeline. GEMMs run on v_wmma_f32_16x16x32_bf16 with
// pre-converted bf16 A operands and fragment-packed bf16 B operands.
// ---------------------------------------------------------------------------

typedef __attribute__((ext_vector_type(16))) __bf16 v16bf;
typedef __attribute__((ext_vector_type(8)))  __bf16 v8bf;
typedef __attribute__((ext_vector_type(8)))  float  v8f;

#define ALPHA 0.2f
#define EPSV  1e-16f
#define GK    256          // shared K for every GEMM in this net
#define KT    (GK / 32)    // 8 k-tiles

// ---------------------------------------------------------------------------
// Pack B[K=256, N] (f32, row-major) into WMMA B-fragment order (bf16):
//   Bp[((kt*ntiles + nt)*32 + lane)*16 + e]  =  B[k(kt,lane,e)][nt*16 + lane%16]
// with k = kt*32 + (lane/16)*8 + (e<8 ? e : 8+e).  Out-of-range N cols -> 0.
// blockIdx.z batches per-head weights.
// ---------------------------------------------------------------------------
__global__ void pack_b_bf16(const float* __restrict__ B, __bf16* __restrict__ Bp,
                            int N, int ntiles, long long srcOff, long long dstOff)
{
    B  += (long long)blockIdx.z * srcOff;
    Bp += (long long)blockIdx.z * dstOff;
    const int idx   = blockIdx.x * blockDim.x + threadIdx.x;
    const int total = KT * ntiles * 512;
    if (idx >= total) return;
    const int e    = idx & 15;
    const int lane = (idx >> 4) & 31;
    const int tile = idx >> 9;
    const int nt   = tile % ntiles;
    const int kt   = tile / ntiles;
    const int n    = nt * 16 + (lane & 15);
    const int k    = kt * 32 + ((lane >> 4) << 3) + (e < 8 ? e : 8 + e);
    const float v  = (n < N) ? B[(size_t)k * N + n] : 0.0f;
    Bp[idx] = (__bf16)v;
}

// ---------------------------------------------------------------------------
// f32 -> bf16 bulk convert
// ---------------------------------------------------------------------------
__global__ void cvt_f32_to_bf16(const float* __restrict__ in,
                                __bf16* __restrict__ out, long long n)
{
    long long i = (long long)blockIdx.x * blockDim.x + threadIdx.x;
    if (i < n) out[i] = (__bf16)in[i];
}

// ---------------------------------------------------------------------------
// WMMA GEMM: C[M,N] = A[M,256]*B[256,N] (+bias).  A bf16 row-major, B packed.
// One wave per 16x16 tile, 8 waves per block, K fully unrolled (8 wmma/tile).
// ---------------------------------------------------------------------------
__global__ __launch_bounds__(256) void gemm_wmma_bf16(
    const __bf16* __restrict__ A, const __bf16* __restrict__ Bp,
    const float* __restrict__ bias, float* __restrict__ C,
    int M, int N, int ldc,
    long long aOff, long long bOff, long long cOff)
{
    A  += (long long)blockIdx.z * aOff;
    Bp += (long long)blockIdx.z * bOff;
    C  += (long long)blockIdx.z * cOff;

    const int ntiles = (N + 15) >> 4;
    const int total  = (M >> 4) * ntiles;
    const int tile   = blockIdx.x * 8 + (threadIdx.x >> 5);
    if (tile >= total) return;                 // wave-uniform exit
    const int mt = tile / ntiles;
    const int nt = tile - mt * ntiles;
    const int m0 = mt << 4;

    const int lane = threadIdx.x & 31;
    const int hi   = lane >> 4;                // 16-lane half
    const int lm   = lane & 15;                // M (A) / N (B,C,D) within tile
    const int kb   = hi << 3;                  // K sub-base 0 or 8

    const __bf16* ap = A  + (size_t)(m0 + lm) * GK + kb;
    const __bf16* bp = Bp + ((size_t)nt * 32 + lane) * 16;
    const size_t  bstep = (size_t)ntiles * 512;   // fragment elems per k-tile

    v8f acc = {};
    #pragma unroll
    for (int kt = 0; kt < KT; ++kt) {
        // A frag: lane row m0+lm, K = k0+kb+{0..7} and k0+kb+16+{0..7}
        v8bf alo = *(const v8bf*)(ap);
        v8bf ahi = *(const v8bf*)(ap + 16);
        v16bf a = __builtin_shufflevector(alo, ahi,
                     0,1,2,3,4,5,6,7,8,9,10,11,12,13,14,15);
        // B frag: one contiguous 32B vector per lane (pre-packed)
        v16bf b = *(const v16bf*)(bp);
        acc = __builtin_amdgcn_wmma_f32_16x16x32_bf16(
                  false, a, false, b, (short)0, acc, false, false);
        ap += 32;
        bp += bstep;
    }

    const int coln = (nt << 4) + lm;
    if (coln < N) {
        const float bv = bias ? bias[coln] : 0.0f;
        #pragma unroll
        for (int r = 0; r < 8; ++r) {          // C/D layout: vgpr r -> M=r(+8)
            const int m = m0 + (hi << 3) + r;
            C[(size_t)m * ldc + coln] = acc[r] + bv;
        }
    }
}

// ---------------------------------------------------------------------------
// Utility: fill buffer with a constant
// ---------------------------------------------------------------------------
__global__ void fill_f32(float* __restrict__ p, float v, long long n)
{
    long long i = (long long)blockIdx.x * blockDim.x + threadIdx.x;
    if (i < n) p[i] = v;
}

// ---------------------------------------------------------------------------
// Per-(node,head) attention scalars: s1 = h.a1, s2 = h.a2 (32-dot each)
// ---------------------------------------------------------------------------
__global__ void head_scalars(const float* __restrict__ h,
                             const float* __restrict__ a1,
                             const float* __restrict__ a2,
                             float* __restrict__ s1, float* __restrict__ s2,
                             int N)
{
    int idx = blockIdx.x * blockDim.x + threadIdx.x;   // n*8 + hd
    if (idx >= N * 8) return;
    const int hd = idx & 7;
    const int n  = idx >> 3;
    const float* hv = h + (size_t)n * 256 + hd * 32;
    const float* w1 = a1 + hd * 32;
    const float* w2 = a2 + hd * 32;
    float d1 = 0.f, d2 = 0.f;
    #pragma unroll
    for (int f = 0; f < 32; ++f) { float v = hv[f]; d1 += v * w1[f]; d2 += v * w2[f]; }
    s1[idx] = d1;
    s2[idx] = d2;
}

// ---------------------------------------------------------------------------
// Per-(edge,head): e = exp(-leaky(s1[row]+s2[col])); scatter e and e*h[col]
// ---------------------------------------------------------------------------
__global__ void gat_edges(const int* __restrict__ row, const int* __restrict__ col,
                          const float* __restrict__ s1, const float* __restrict__ s2,
                          const float* __restrict__ h,
                          float* __restrict__ rowsum, float* __restrict__ hp, int E)
{
    int idx = blockIdx.x * blockDim.x + threadIdx.x;   // e*8 + hd
    if (idx >= E * 8) return;
    const int hd = idx & 7;
    const int e  = idx >> 3;
    const int r = row[e], c = col[e];
    const float lg = s1[r * 8 + hd] + s2[c * 8 + hd];
    const float lr = lg > 0.f ? lg : ALPHA * lg;
    const float w  = __expf(-lr);
    atomicAdd(&rowsum[r * 8 + hd], w);
    const float* hc  = h  + (size_t)c * 256 + hd * 32;
    float*       out = hp + (size_t)r * 256 + hd * 32;
    #pragma unroll
    for (int f = 0; f < 32; f += 4) {
        float4 v = *(const float4*)(hc + f);
        atomicAdd(out + f + 0, w * v.x);
        atomicAdd(out + f + 1, w * v.y);
        atomicAdd(out + f + 2, w * v.z);
        atomicAdd(out + f + 3, w * v.w);
    }
}

// ---------------------------------------------------------------------------
// h1 = elu(hp / (rowsum + eps)) -> bf16 (GEMM operand layout, row-major)
// ---------------------------------------------------------------------------
__global__ void normalize_elu_bf16(const float* __restrict__ hp,
                                   const float* __restrict__ rowsum,
                                   __bf16* __restrict__ out, int N)
{
    int idx = blockIdx.x * blockDim.x + threadIdx.x;   // n*256 + c
    if (idx >= N * 256) return;
    const int n  = idx >> 8;
    const int c  = idx & 255;
    const int hd = c >> 5;
    float v = hp[idx] / (rowsum[n * 8 + hd] + EPSV);
    v = v > 0.f ? v : __expf(v) - 1.f;
    out[idx] = (__bf16)v;
}

// ---------------------------------------------------------------------------
// h2 init = b_gc broadcast
// ---------------------------------------------------------------------------
__global__ void gc_init(float* __restrict__ h2, const float* __restrict__ b, int N)
{
    int idx = blockIdx.x * blockDim.x + threadIdx.x;
    if (idx >= N * 256) return;
    h2[idx] = b[idx & 255];
}

// ---------------------------------------------------------------------------
// GraphConv aggregation: h2[row[e]] += support[col[e]]  (1 block == 1 edge)
// ---------------------------------------------------------------------------
__global__ __launch_bounds__(256) void gc_aggregate(
    const int* __restrict__ row, const int* __restrict__ col,
    const float* __restrict__ support, float* __restrict__ h2)
{
    const int e = blockIdx.x;
    const int f = threadIdx.x;
    const int r = row[e], c = col[e];
    atomicAdd(&h2[(size_t)r * 256 + f], support[(size_t)c * 256 + f]);
}

// ---------------------------------------------------------------------------
// Output-head attention scalars (40-dot)
// ---------------------------------------------------------------------------
__global__ void out_scalars(const float* __restrict__ h,
                            const float* __restrict__ a1,
                            const float* __restrict__ a2,
                            float* __restrict__ s1, float* __restrict__ s2, int N)
{
    int n = blockIdx.x * blockDim.x + threadIdx.x;
    if (n >= N) return;
    const float* hv = h + (size_t)n * 40;
    float d1 = 0.f, d2 = 0.f;
    #pragma unroll
    for (int f = 0; f < 40; ++f) { float v = hv[f]; d1 += v * a1[f]; d2 += v * a2[f]; }
    s1[n] = d1;
    s2[n] = d2;
}

// ---------------------------------------------------------------------------
// Output-head edges: scatter e and e*h_out[col]
// ---------------------------------------------------------------------------
__global__ void out_edges(const int* __restrict__ row, const int* __restrict__ col,
                          const float* __restrict__ s1, const float* __restrict__ s2,
                          const float* __restrict__ h,
                          float* __restrict__ rs, float* __restrict__ hp, int E)
{
    int e = blockIdx.x * blockDim.x + threadIdx.x;
    if (e >= E) return;
    const int r = row[e], c = col[e];
    const float lg = s1[r] + s2[c];
    const float lr = lg > 0.f ? lg : ALPHA * lg;
    const float w  = __expf(-lr);
    atomicAdd(&rs[r], w);
    const float* hc  = h  + (size_t)c * 40;
    float*       out = hp + (size_t)r * 40;
    #pragma unroll
    for (int f = 0; f < 40; ++f) atomicAdd(out + f, w * hc[f]);
}

// ---------------------------------------------------------------------------
// xo = elu(hp/(rs+eps)); d_out[n,:] = log_softmax(xo) over 40 classes
// ---------------------------------------------------------------------------
__global__ void finalize_logsoftmax(const float* __restrict__ hp,
                                    const float* __restrict__ rs,
                                    float* __restrict__ out, int N)
{
    int n = blockIdx.x * blockDim.x + threadIdx.x;
    if (n >= N) return;
    const float inv = 1.0f / (rs[n] + EPSV);
    float v[40];
    float mx = -1e30f;
    #pragma unroll
    for (int f = 0; f < 40; ++f) {
        float t = hp[(size_t)n * 40 + f] * inv;
        t = t > 0.f ? t : __expf(t) - 1.f;     // elu
        v[f] = t;
        mx = fmaxf(mx, t);
    }
    float s = 0.f;
    #pragma unroll
    for (int f = 0; f < 40; ++f) s += __expf(v[f] - mx);
    const float ls = __logf(s);
    #pragma unroll
    for (int f = 0; f < 40; ++f) out[(size_t)n * 40 + f] = v[f] - mx - ls;
}

// ---------------------------------------------------------------------------
// Launch
// ---------------------------------------------------------------------------
extern "C" void kernel_launch(void* const* d_in, const int* in_sizes, int n_in,
                              void* d_out, int out_size, void* d_ws, size_t ws_size,
                              hipStream_t stream)
{
    (void)n_in; (void)out_size; (void)ws_size;

    const float* x     = (const float*)d_in[0];
    const int*   eidx  = (const int*)  d_in[1];
    const float* W_att = (const float*)d_in[2];
    const float* a1    = (const float*)d_in[3];
    const float* a2    = (const float*)d_in[4];
    const float* W_out = (const float*)d_in[5];
    const float* a1o   = (const float*)d_in[6];
    const float* a2o   = (const float*)d_in[7];
    const float* W_gc  = (const float*)d_in[8];
    const float* b_gc  = (const float*)d_in[9];
    const float* W_enc = (const float*)d_in[10];
    const float* b_enc = (const float*)d_in[11];

    const int N = in_sizes[0] / 256;   // 50000
    const int E = in_sizes[1] / 2;     // 800000
    const int* erow = eidx;
    const int* ecol = eidx + E;

    // workspace layout: all region sizes are multiples of 32 bytes, so the
    // packed-bf16 vector loads stay 32B-aligned given a 256B-aligned d_ws.
    char* base = (char*)d_ws;
    auto alloc_f = [&](size_t n) { float*  p = (float*)base;  base += n * 4; return p; };
    auto alloc_h = [&](size_t n) { __bf16* p = (__bf16*)base; base += n * 2; return p; };

    float*  h_heads = alloc_f((size_t)N * 256);  // x @ W_att (f32, head-major cols)
    float*  hp1     = alloc_f((size_t)N * 256);  // attention accumulator
    float*  s1      = alloc_f((size_t)N * 8);
    float*  s2      = alloc_f((size_t)N * 8);
    float*  rowsum  = alloc_f((size_t)N * 8);
    float*  h2      = alloc_f((size_t)N * 256);
    float*  support = alloc_f((size_t)N * 256);
    float*  h_out   = alloc_f((size_t)N * 40);
    float*  s1o     = alloc_f(N);
    float*  s2o     = alloc_f(N);
    float*  rso     = alloc_f(N);
    float*  hpo     = alloc_f((size_t)N * 40);
    __bf16* xbf     = alloc_h((size_t)N * 256);  // bf16 GEMM A operands
    __bf16* h1bf    = alloc_h((size_t)N * 256);
    __bf16* h2bf    = alloc_h((size_t)N * 256);
    __bf16* wattP   = alloc_h(8 * KT * 2  * 512);  // packed B fragments
    __bf16* wgcP    = alloc_h(KT * 16 * 512);
    __bf16* woutP   = alloc_h(KT * 3  * 512);
    __bf16* wencP   = alloc_h(KT * 8  * 512);

    float* out_ls = (float*)d_out;                 // [N,40]
    float* out_y  = out_ls + (size_t)N * 40;       // [N,128]
    float* out_z  = out_y  + (size_t)N * 128;      // [N,128]

    const int TPB = 256;
    auto blks = [](long long n, int t) { return (int)((n + t - 1) / t); };

    // 0) pack weights into WMMA fragment order; convert x to bf16
    pack_b_bf16<<<dim3(blks(KT * 2 * 512, TPB), 1, 8), TPB, 0, stream>>>(
        W_att, wattP, 32, 2, 256LL * 32, (long long)KT * 2 * 512);
    pack_b_bf16<<<dim3(blks(KT * 16 * 512, TPB)), TPB, 0, stream>>>(W_gc,  wgcP,  256, 16, 0, 0);
    pack_b_bf16<<<dim3(blks(KT * 3  * 512, TPB)), TPB, 0, stream>>>(W_out, woutP, 40,  3,  0, 0);
    pack_b_bf16<<<dim3(blks(KT * 8  * 512, TPB)), TPB, 0, stream>>>(W_enc, wencP, 128, 8,  0, 0);
    cvt_f32_to_bf16<<<blks((long long)N * 256, TPB), TPB, 0, stream>>>(x, xbf, (long long)N * 256);

    // 1) h_heads = x @ W_att  (8 heads over z; head h -> column block h*32)
    {
        int tiles = (N / 16) * 2;
        gemm_wmma_bf16<<<dim3(blks(tiles, 8), 1, 8), TPB, 0, stream>>>(
            xbf, wattP, nullptr, h_heads, N, 32, 256,
            0LL, (long long)KT * 2 * 512, 32LL);
    }
    // 2) attention scalars
    head_scalars<<<blks((long long)N * 8, TPB), TPB, 0, stream>>>(h_heads, a1, a2, s1, s2, N);
    // 3) zero accumulators
    fill_f32<<<blks((long long)N * 256, TPB), TPB, 0, stream>>>(hp1, 0.f, (long long)N * 256);
    fill_f32<<<blks((long long)N * 8, TPB), TPB, 0, stream>>>(rowsum, 0.f, (long long)N * 8);
    // 4) sparse attention edges (8 heads)
    gat_edges<<<blks((long long)E * 8, TPB), TPB, 0, stream>>>(erow, ecol, s1, s2,
                                                               h_heads, rowsum, hp1, E);
    // 5) normalize + elu -> h1 (bf16)
    normalize_elu_bf16<<<blks((long long)N * 256, TPB), TPB, 0, stream>>>(hp1, rowsum, h1bf, N);
    // 6) support = h1 @ W_gc
    gemm_wmma_bf16<<<dim3(blks((N / 16) * 16, 8)), TPB, 0, stream>>>(
        h1bf, wgcP, nullptr, support, N, 256, 256, 0, 0, 0);
    // 7) h2 = b_gc; h2[row] += support[col]; convert to bf16
    gc_init<<<blks((long long)N * 256, TPB), TPB, 0, stream>>>(h2, b_gc, N);
    gc_aggregate<<<dim3(E), 256, 0, stream>>>(erow, ecol, support, h2);
    cvt_f32_to_bf16<<<blks((long long)N * 256, TPB), TPB, 0, stream>>>(h2, h2bf, (long long)N * 256);
    // 8) h_out = h2 @ W_out  (N=40: B zero-padded at pack time, stores guarded)
    gemm_wmma_bf16<<<dim3(blks((N / 16) * 3, 8)), TPB, 0, stream>>>(
        h2bf, woutP, nullptr, h_out, N, 40, 40, 0, 0, 0);
    // 9) output attention head (concat=False)
    out_scalars<<<blks(N, TPB), TPB, 0, stream>>>(h_out, a1o, a2o, s1o, s2o, N);
    fill_f32<<<blks(N, TPB), TPB, 0, stream>>>(rso, 0.f, N);
    fill_f32<<<blks((long long)N * 40, TPB), TPB, 0, stream>>>(hpo, 0.f, (long long)N * 40);
    out_edges<<<blks(E, TPB), TPB, 0, stream>>>(erow, ecol, s1o, s2o, h_out, rso, hpo, E);
    // 10) elu + log_softmax -> d_out[0:N*40]
    finalize_logsoftmax<<<blks(N, TPB), TPB, 0, stream>>>(hpo, rso, out_ls, N);
    // 11) y = h1 @ W_enc + b_enc ; z = h2 @ W_enc + b_enc
    gemm_wmma_bf16<<<dim3(blks((N / 16) * 8, 8)), TPB, 0, stream>>>(
        h1bf, wencP, b_enc, out_y, N, 128, 128, 0, 0, 0);
    gemm_wmma_bf16<<<dim3(blks((N / 16) * 8, 8)), TPB, 0, stream>>>(
        h2bf, wencP, b_enc, out_z, N, 128, 128, 0, 0, 0);
}
